// MPNNSurrogate_38886633898629
// MI455X (gfx1250) — matrix operand order repaired and checked by reading, not compiled, not run on hardware
//
#include <hip/hip_runtime.h>
#include <hip/hip_bf16.h>

#define NNODE  50000
#define NEDGE  800000
#define HDIM   64
#define NLAYER 3
#define EPSV   1e-5f

typedef __attribute__((ext_vector_type(16))) __bf16 v16bf;
typedef __attribute__((ext_vector_type(8)))  __bf16 v8bf;
typedef __attribute__((ext_vector_type(8)))  float  v8f;

__device__ __forceinline__ __bf16 tobf(float f) { return (__bf16)f; }

__device__ __forceinline__ v8f vzero() {
  v8f z;
  #pragma unroll
  for (int i = 0; i < 8; i++) z[i] = 0.f;
  return z;
}

__device__ __forceinline__ v8f wmma_bf16(v16bf a, v16bf b, v8f c) {
#if defined(__gfx1250__)
  // D(16x16,f32) = A(16x32,bf16) x B(32x16,bf16) + C
  return __builtin_amdgcn_wmma_f32_16x16x32_bf16(false, a, false, b, (short)0, c,
                                                 false, false);
#else
  (void)a; (void)b;
  return c;
#endif
}

// Convert a K x 64 fp32 weight matrix into bf16 "B-fragment order" in LDS:
// flat index i = ((kt*4 + ntile)*32 + lane)*16 + e,
// value = W[(kt*32 + (e&8)*2 + (lane>=16?8:0) + (e&7)) * 64 + (ntile*16 + (lane&15))].
// Each B fragment is then one contiguous 32B per-lane LDS read.
__device__ __forceinline__ void build_wfrag(const float* __restrict__ W,
                                            __bf16* frag, int Kdim,
                                            int tid, int nth) {
  int total = Kdim * 64;
  for (int i = tid; i < total; i += nth) {
    int e  = i & 15;
    int l  = (i >> 4) & 31;
    int ft = (i >> 9) & 3;
    int kt = i >> 11;
    int k  = kt * 32 + (e & 8) * 2 + ((l >= 16) ? 8 : 0) + (e & 7);
    int n  = ft * 16 + (l & 15);
    frag[i] = tobf(W[k * 64 + n]);
  }
}

// A fragment (16x32 bf16) from LDS row-major tile (lda in bf16 units, lda%8==0).
// ISA layout: element e of lane -> K(e) = (e&8)*2 + hi*8 + (e&7), row = lane&15.
__device__ __forceinline__ v16bf load_afrag(const __bf16* As, int lda, int kt,
                                            int lane) {
  int row = lane & 15;
  int hi  = (lane >> 4) & 1;
  const __bf16* p = As + row * lda + kt * 32 + hi * 8;
  v8bf lo = *(const v8bf*)p;
  v8bf hh = *(const v8bf*)(p + 16);
  v16bf a;
  #pragma unroll
  for (int i = 0; i < 8; i++) { a[i] = lo[i]; a[8 + i] = hh[i]; }
  return a;
}

// One wave computes D(16x64) += A(16x(32*KT)) @ W((32*KT)x64).
template <int KT>
__device__ __forceinline__ void gemm_acc(const __bf16* As, int lda,
                                         const __bf16* Wf, v8f acc[4],
                                         int lane) {
  #pragma unroll
  for (int kt = 0; kt < KT; kt++) {
    v16bf a = load_afrag(As, lda, kt, lane);
    #pragma unroll
    for (int t = 0; t < 4; t++) {
      v16bf b = *(const v16bf*)(Wf + (((kt * 4 + t) * 32 + lane) << 4));
      acc[t] = wmma_bf16(a, b, acc[t]);
    }
  }
}

// Store 16x64 accumulators (+bias) into per-wave fp32 LDS stage (stride 64).
__device__ __forceinline__ void store_acc_f32(const v8f acc[4], float* sf,
                                              const float* __restrict__ bias,
                                              int lane) {
  int cl = lane & 15, hi = lane >> 4;
  #pragma unroll
  for (int t = 0; t < 4; t++)
    #pragma unroll
    for (int r = 0; r < 8; r++) {
      int col = t * 16 + cl, row = r + hi * 8;
      sf[row * 64 + col] = acc[t][r] + bias[col];
    }
}

// Per-row LayerNorm (fp32) + ReLU, result into bf16 stage (stride 72).
// sb may alias sf (reads precede overlapping writes in program order).
__device__ __forceinline__ void ln_relu_bf16(const float* sf, __bf16* sb,
                                             const float* __restrict__ gv,
                                             const float* __restrict__ bev,
                                             float* mrow, float* rrow,
                                             int lane) {
  if (lane < 16) {
    const float* rp = sf + lane * 64;
    float m = 0.f;
    #pragma unroll 8
    for (int c = 0; c < 64; c++) m += rp[c];
    m *= (1.f / 64.f);
    float v = 0.f;
    #pragma unroll 8
    for (int c = 0; c < 64; c++) { float d = rp[c] - m; v += d * d; }
    v *= (1.f / 64.f);
    mrow[lane] = m;
    rrow[lane] = rsqrtf(v + EPSV);
  }
  for (int idx = lane; idx < 16 * 64; idx += 32) {
    int row = idx >> 6, col = idx & 63;
    float val = (sf[idx] - mrow[row]) * rrow[row] * gv[col] + bev[col];
    sb[row * 72 + col] = tobf(fmaxf(val, 0.f));
  }
}

// ---------------- encoder (node: DIN=2, edge: DIN=5) ----------------
// Persistent waves: weight fragments built once per block, then each wave
// grid-strides over 16-row tiles. launch_bounds(.,4): cap VGPRs at 256.
template <int DIN, int EDGE>
__global__ __launch_bounds__(256, 4) void encoder_kernel(
    const float* __restrict__ X,
    const float* __restrict__ w1, const float* __restrict__ b1,
    const float* __restrict__ g1, const float* __restrict__ be1,
    const float* __restrict__ w2, const float* __restrict__ b2,
    const float* __restrict__ g2, const float* __restrict__ be2,
    const float* __restrict__ w3, const float* __restrict__ b3,
    float* __restrict__ node_out, __bf16* __restrict__ edge_out,
    float* __restrict__ edge_sum, int ntiles) {
  __shared__ __attribute__((aligned(32))) __bf16 w2f[64 * 64];
  __shared__ __attribute__((aligned(32))) __bf16 w3f[64 * 64];
  __shared__ __attribute__((aligned(32))) float stage[8][16 * 64];
  __shared__ float mr[8][16], rr[8][16];

  int tid = threadIdx.x, lane = tid & 31, w = tid >> 5;
  build_wfrag(w2, w2f, 64, tid, 256);
  build_wfrag(w3, w3f, 64, tid, 256);
  __syncthreads();

  float*  sf = stage[w];
  __bf16* sb = (__bf16*)sf;  // aliased bf16 stage (stride 72)
  int cl = lane & 15, hi = lane >> 4;
  int wavesTotal = gridDim.x * 8;

  #pragma unroll 1
  for (int tile = blockIdx.x * 8 + w; tile < ntiles; tile += wavesTotal) {
    int rbase = tile * 16;

    // layer 1: tiny K -> plain VALU fp32
    for (int idx = lane; idx < 16 * 64; idx += 32) {
      int row = idx >> 6, col = idx & 63;
      const float* xr = X + (size_t)(rbase + row) * DIN;
      float a = b1[col];
      #pragma unroll
      for (int k = 0; k < DIN; k++) a += xr[k] * w1[k * 64 + col];
      sf[idx] = a;
    }
    ln_relu_bf16(sf, sb, g1, be1, mr[w], rr[w], lane);

    v8f acc[4];
    #pragma unroll
    for (int t = 0; t < 4; t++) acc[t] = vzero();
    gemm_acc<2>(sb, 72, w2f, acc, lane);
    store_acc_f32(acc, sf, b2, lane);
    ln_relu_bf16(sf, sb, g2, be2, mr[w], rr[w], lane);

    #pragma unroll
    for (int t = 0; t < 4; t++) acc[t] = vzero();
    gemm_acc<2>(sb, 72, w3f, acc, lane);

    if (EDGE) {
      #pragma unroll
      for (int t = 0; t < 4; t++) {
        float cs = 0.f;
        #pragma unroll
        for (int r = 0; r < 8; r++) {
          int col = t * 16 + cl, row = r + hi * 8;
          float v = acc[t][r] + b3[col];
          edge_out[(size_t)(rbase + row) * 64 + col] = tobf(v);
          cs += v;
        }
        cs += __shfl_down(cs, 16, 32);  // fold rows 8..15 into lanes 0..15
        if (lane < 16) atomicAdd(&edge_sum[t * 16 + cl], cs);
      }
    } else {
      #pragma unroll
      for (int t = 0; t < 4; t++)
        #pragma unroll
        for (int r = 0; r < 8; r++) {
          int col = t * 16 + cl, row = r + hi * 8;
          node_out[(size_t)(rbase + row) * 64 + col] = acc[t][r] + b3[col];
        }
    }
  }
}

// ---------------- message MLP + fused segment-sum scatter ----------------
__global__ __launch_bounds__(128, 4) void msg_kernel(
    const float* __restrict__ node_h, const __bf16* __restrict__ edge_h,
    const int* __restrict__ src, const int* __restrict__ tgt,
    const float* __restrict__ mw1, const float* __restrict__ mb1,
    const float* __restrict__ mw2, const float* __restrict__ mb2,
    float* __restrict__ agg, int ntiles) {
  __shared__ __attribute__((aligned(32))) __bf16 w1f[192 * 64];
  __shared__ __attribute__((aligned(32))) __bf16 w2f[64 * 64];
  __shared__ __attribute__((aligned(32))) __bf16 astage[4][16 * 200];
  __shared__ int eidx_s[4][32];

  int tid = threadIdx.x, lane = tid & 31, w = tid >> 5;
  build_wfrag(mw1, w1f, 192, tid, 128);
  build_wfrag(mw2, w2f, 64, tid, 128);
  __syncthreads();

  __bf16* as = astage[w];
  int cl = lane & 15, hi = lane >> 4;
  int wavesTotal = gridDim.x * 4;

  #pragma unroll 1
  for (int tile = blockIdx.x * 4 + w; tile < ntiles; tile += wavesTotal) {
    int ebase = tile * 16;

    if (lane < 16) {
      eidx_s[w][lane]      = src[ebase + lane];
      eidx_s[w][16 + lane] = tgt[ebase + lane];
    }

    // stage A = [node_h[src] | node_h[tgt] | edge_h]  (16 x 192, stride 200)
    for (int idx = lane; idx < 16 * 192; idx += 32) {
      int r = idx / 192, c = idx % 192;
      float v;
      if (c < 64)       v = node_h[(size_t)eidx_s[w][r] * 64 + c];
      else if (c < 128) v = node_h[(size_t)eidx_s[w][16 + r] * 64 + (c - 64)];
      else              v = (float)edge_h[(size_t)(ebase + r) * 64 + (c - 128)];
      as[r * 200 + c] = tobf(v);
    }

    v8f acc[4];
    #pragma unroll
    for (int t = 0; t < 4; t++) acc[t] = vzero();
    gemm_acc<6>(as, 200, w1f, acc, lane);

    __bf16* hb = as;  // hidden stage aliases A stage (stride 72)
    #pragma unroll
    for (int t = 0; t < 4; t++)
      #pragma unroll
      for (int r = 0; r < 8; r++) {
        int col = t * 16 + cl, row = r + hi * 8;
        hb[row * 72 + col] = tobf(fmaxf(acc[t][r] + mb1[col], 0.f));
      }

    #pragma unroll
    for (int t = 0; t < 4; t++) acc[t] = vzero();
    gemm_acc<2>(hb, 72, w2f, acc, lane);

    // scatter msgs into agg[tgt] (segment_sum)
    #pragma unroll
    for (int t = 0; t < 4; t++)
      #pragma unroll
      for (int r = 0; r < 8; r++) {
        int col = t * 16 + cl, row = r + hi * 8;
        int tg  = eidx_s[w][16 + row];
        atomicAdd(&agg[(size_t)tg * 64 + col], acc[t][r] + mb2[col]);
      }
  }
}

// ---------------- node update MLP (+ residual, in place) ----------------
__global__ __launch_bounds__(256, 4) void update_kernel(
    float* __restrict__ node_h, const float* __restrict__ agg,
    const float* __restrict__ uw1, const float* __restrict__ ub1,
    const float* __restrict__ uw2, const float* __restrict__ ub2, int ntiles) {
  __shared__ __attribute__((aligned(32))) __bf16 u1f[128 * 64];
  __shared__ __attribute__((aligned(32))) __bf16 u2f[64 * 64];
  __shared__ __attribute__((aligned(32))) __bf16 astage[8][16 * 136];

  int tid = threadIdx.x, lane = tid & 31, w = tid >> 5;
  build_wfrag(uw1, u1f, 128, tid, 256);
  build_wfrag(uw2, u2f, 64, tid, 256);
  __syncthreads();

  __bf16* as = astage[w];
  int cl = lane & 15, hi = lane >> 4;
  int wavesTotal = gridDim.x * 8;

  #pragma unroll 1
  for (int tile = blockIdx.x * 8 + w; tile < ntiles; tile += wavesTotal) {
    int rbase = tile * 16;

    for (int idx = lane; idx < 16 * 128; idx += 32) {
      int r = idx >> 7, c = idx & 127;
      float v = (c < 64) ? node_h[(size_t)(rbase + r) * 64 + c]
                         : agg[(size_t)(rbase + r) * 64 + (c - 64)];
      as[r * 136 + c] = tobf(v);
    }

    v8f acc[4];
    #pragma unroll
    for (int t = 0; t < 4; t++) acc[t] = vzero();
    gemm_acc<4>(as, 136, u1f, acc, lane);

    __bf16* hb = as;  // alias (stride 72)
    #pragma unroll
    for (int t = 0; t < 4; t++)
      #pragma unroll
      for (int r = 0; r < 8; r++) {
        int col = t * 16 + cl, row = r + hi * 8;
        hb[row * 72 + col] = tobf(fmaxf(acc[t][r] + ub1[col], 0.f));
      }

    #pragma unroll
    for (int t = 0; t < 4; t++) acc[t] = vzero();
    gemm_acc<2>(hb, 72, u2f, acc, lane);

    #pragma unroll
    for (int t = 0; t < 4; t++)
      #pragma unroll
      for (int r = 0; r < 8; r++) {
        int col = t * 16 + cl, row = r + hi * 8;
        size_t gi = (size_t)(rbase + row) * 64 + col;
        node_h[gi] = node_h[gi] + acc[t][r] + ub2[col];
      }
  }
}

// ---------------- utilities ----------------
__global__ void zero_kernel(float* p, int n) {
  int i = blockIdx.x * blockDim.x + threadIdx.x;
  if (i < n) p[i] = 0.f;
}

__global__ __launch_bounds__(256) void pool_kernel(const float* __restrict__ node_h,
                                                   float* __restrict__ node_sum) {
  __shared__ float ps[64];
  int tid = threadIdx.x;
  if (tid < 64) ps[tid] = 0.f;
  __syncthreads();
  int total = NNODE * 64;
  for (int i = blockIdx.x * blockDim.x + tid; i < total;
       i += gridDim.x * blockDim.x)
    atomicAdd(&ps[i & 63], node_h[i]);
  __syncthreads();
  if (tid < 64) atomicAdd(&node_sum[tid], ps[tid]);
}

__global__ __launch_bounds__(64) void head_kernel(
    const float* __restrict__ node_sum, const float* __restrict__ edge_sum,
    const float* __restrict__ w1, const float* __restrict__ b1,
    const float* __restrict__ g1, const float* __restrict__ be1,
    const float* __restrict__ w2, const float* __restrict__ b2,
    const float* __restrict__ g2, const float* __restrict__ be2,
    const float* __restrict__ w3, const float* __restrict__ b3,
    float* __restrict__ out) {
  __shared__ float gg[128];
  __shared__ float buf[64];
  __shared__ float hbuf[64];
  int j = threadIdx.x;
  gg[j]      = node_sum[j] * (1.f / (float)NNODE);
  gg[64 + j] = edge_sum[j] * (1.f / (float)NEDGE);
  __syncthreads();

  float a = b1[j];
  for (int k = 0; k < 128; k++) a += gg[k] * w1[k * 64 + j];
  buf[j] = a;
  __syncthreads();
  float m = 0.f, v = 0.f;
  for (int k = 0; k < 64; k++) m += buf[k];
  m *= (1.f / 64.f);
  for (int k = 0; k < 64; k++) { float d = buf[k] - m; v += d * d; }
  v *= (1.f / 64.f);
  float h = fmaxf((a - m) * rsqrtf(v + EPSV) * g1[j] + be1[j], 0.f);
  hbuf[j] = h;
  __syncthreads();

  a = b2[j];
  for (int k = 0; k < 64; k++) a += hbuf[k] * w2[k * 64 + j];
  __syncthreads();
  buf[j] = a;
  __syncthreads();
  m = 0.f; v = 0.f;
  for (int k = 0; k < 64; k++) m += buf[k];
  m *= (1.f / 64.f);
  for (int k = 0; k < 64; k++) { float d = buf[k] - m; v += d * d; }
  v *= (1.f / 64.f);
  h = fmaxf((a - m) * rsqrtf(v + EPSV) * g2[j] + be2[j], 0.f);
  __syncthreads();
  buf[j] = h * w3[j];
  __syncthreads();
  if (j == 0) {
    float y = b3[0];
    for (int k = 0; k < 64; k++) y += buf[k];
    out[0] = (y > 20.f) ? y : log1pf(expf(y));  // softplus
  }
}

extern "C" void kernel_launch(void* const* d_in, const int* in_sizes, int n_in,
                              void* d_out, int out_size, void* d_ws,
                              size_t ws_size, hipStream_t stream) {
  (void)in_sizes; (void)n_in; (void)out_size; (void)ws_size;
  const float* node_feats = (const float*)d_in[0];
  const float* edge_feats = (const float*)d_in[1];
  const int*   eidx       = (const int*)d_in[2];
  const int* esrc = eidx;
  const int* etgt = eidx + NEDGE;
  const float* ne[10]; for (int i = 0; i < 10; i++) ne[i] = (const float*)d_in[3 + i];
  const float* ee[10]; for (int i = 0; i < 10; i++) ee[i] = (const float*)d_in[13 + i];
  const float* mp_mw1 = (const float*)d_in[23];
  const float* mp_mb1 = (const float*)d_in[24];
  const float* mp_mw2 = (const float*)d_in[25];
  const float* mp_mb2 = (const float*)d_in[26];
  const float* mp_uw1 = (const float*)d_in[27];
  const float* mp_ub1 = (const float*)d_in[28];
  const float* mp_uw2 = (const float*)d_in[29];
  const float* mp_ub2 = (const float*)d_in[30];
  const float* hm[10]; for (int i = 0; i < 10; i++) hm[i] = (const float*)d_in[31 + i];

  // workspace carve (~128.5 MB; fits in the 192 MB L2 alongside node_h reuse)
  char* ws = (char*)d_ws;
  size_t off = 0;
  float* node_h = (float*)(ws + off);
  off += (size_t)NNODE * HDIM * sizeof(float); off = (off + 255) & ~(size_t)255;
  float* agg = (float*)(ws + off);
  off += (size_t)NNODE * HDIM * sizeof(float); off = (off + 255) & ~(size_t)255;
  __bf16* edge_h = (__bf16*)(ws + off);
  off += (size_t)NEDGE * HDIM * sizeof(__bf16); off = (off + 255) & ~(size_t)255;
  float* node_sum = (float*)(ws + off);             // 64 floats
  float* edge_sum = (float*)(ws + off) + 64;        // 64 floats, contiguous

  const int ntilesN = NNODE / 16;   // 3125
  const int ntilesE = NEDGE / 16;   // 50000

  zero_kernel<<<1, 128, 0, stream>>>(node_sum, 128);  // node_sum + edge_sum

  // persistent-wave grids: amortize weight-fragment build over many tiles
  encoder_kernel<2, 0><<<200, 256, 0, stream>>>(
      node_feats, ne[0], ne[1], ne[2], ne[3], ne[4], ne[5], ne[6], ne[7], ne[8],
      ne[9], node_h, (__bf16*)nullptr, (float*)nullptr, ntilesN);

  encoder_kernel<5, 1><<<1250, 256, 0, stream>>>(
      edge_feats, ee[0], ee[1], ee[2], ee[3], ee[4], ee[5], ee[6], ee[7], ee[8],
      ee[9], (float*)nullptr, edge_h, edge_sum, ntilesE);

  for (int l = 0; l < NLAYER; l++) {
    zero_kernel<<<(NNODE * HDIM + 255) / 256, 256, 0, stream>>>(agg, NNODE * HDIM);
    msg_kernel<<<2500, 128, 0, stream>>>(
        node_h, edge_h, esrc, etgt,
        mp_mw1 + (size_t)l * 3 * HDIM * HDIM, mp_mb1 + l * HDIM,
        mp_mw2 + (size_t)l * HDIM * HDIM, mp_mb2 + l * HDIM, agg, ntilesE);
    update_kernel<<<200, 256, 0, stream>>>(
        node_h, agg,
        mp_uw1 + (size_t)l * 2 * HDIM * HDIM, mp_ub1 + l * HDIM,
        mp_uw2 + (size_t)l * HDIM * HDIM, mp_ub2 + l * HDIM, ntilesN);
  }

  pool_kernel<<<256, 256, 0, stream>>>(node_h, node_sum);
  head_kernel<<<1, 64, 0, stream>>>(node_sum, edge_sum, hm[0], hm[1], hm[2],
                                    hm[3], hm[4], hm[5], hm[6], hm[7], hm[8],
                                    hm[9], (float*)d_out);
}
